// ClusteringLayer_35570919145510
// MI455X (gfx1250) — matrix operand order repaired and checked by reading, not compiled, not run on hardware
//
#include <hip/hip_runtime.h>
#include <hip/hip_bf16.h>
#include <stdint.h>

typedef __attribute__((ext_vector_type(16))) __bf16 v16bf;
typedef __attribute__((ext_vector_type(8)))  float  v8f;

#define D_DIM 256
#define K_DIM 256
#define ROWS_PER_WG 128
#define LDS_STRIDE 264   // bf16 elements per padded row (256 + 8) -> 528B, 16B aligned

union BFrag { v16bf v; uint4 u[2]; };

__device__ __forceinline__ uint32_t pack2bf(float a, float b) {
    __bf16 ha = (__bf16)a, hb = (__bf16)b;
    uint16_t ua = __builtin_bit_cast(uint16_t, ha);
    uint16_t ub = __builtin_bit_cast(uint16_t, hb);
    return ((uint32_t)ub << 16) | (uint32_t)ua;
}

// ---- one-shot prep: C (256x256 f32) -> bf16 in ws, plus ||c||^2 + 1 ----
__global__ __launch_bounds__(256)
void prep_kernel(const float* __restrict__ C,
                 __bf16* __restrict__ Cbf,      // 256*256 bf16, packed row-major
                 float* __restrict__ csq1)      // 256 floats
{
    const int tid = threadIdx.x;               // one cluster row per thread
    const float4* crow = (const float4*)(C + (size_t)tid * D_DIM);
    uint4* dst = (uint4*)(Cbf + (size_t)tid * D_DIM);
    float s = 1.0f;
    #pragma unroll
    for (int i = 0; i < D_DIM / 8; ++i) {
        float4 f0 = crow[2*i], f1 = crow[2*i+1];
        s += f0.x*f0.x + f0.y*f0.y + f0.z*f0.z + f0.w*f0.w;
        s += f1.x*f1.x + f1.y*f1.y + f1.z*f1.z + f1.w*f1.w;
        uint4 pk;
        pk.x = pack2bf(f0.x, f0.y); pk.y = pack2bf(f0.z, f0.w);
        pk.z = pack2bf(f1.x, f1.y); pk.w = pack2bf(f1.z, f1.w);
        dst[i] = pk;
    }
    csq1[tid] = s;
}

__global__ __launch_bounds__(256)
void cluster_q_kernel(const float* __restrict__ X,
                      const __bf16* __restrict__ Cbf,
                      const float* __restrict__ csq1,
                      float* __restrict__ Q,
                      int N)
{
    __shared__ __bf16 Cs[K_DIM * LDS_STRIDE];   // clusters, bf16, padded rows
    __shared__ float  csq1_s[K_DIM];            // ||c||^2 + 1

    const int tid  = threadIdx.x;
    const int lane = tid & 31;
    const int wave = tid >> 5;
    const int hf   = lane >> 4;        // 0 = lanes 0-15, 1 = lanes 16-31
    const int l16  = lane & 15;

    // ---- async-DMA the pre-converted bf16 clusters into padded LDS rows ----
    {
        // one 512B cluster row per thread -> padded 528B LDS row (32 x b128)
        uint32_t lbase = (uint32_t)(uintptr_t)(&Cs[0]) + (uint32_t)tid * (LDS_STRIDE * 2);
        const char* gbase = (const char*)Cbf + (size_t)tid * (D_DIM * 2);
        #pragma unroll
        for (int i = 0; i < (D_DIM * 2) / 16; ++i) {
            uint32_t laddr = lbase + i * 16;
            uint64_t gaddr = (uint64_t)(uintptr_t)(gbase + i * 16);
            asm volatile("global_load_async_to_lds_b128 %0, %1, off"
                         :: "v"(laddr), "v"(gaddr) : "memory");
        }
        csq1_s[tid] = csq1[tid];
        asm volatile("s_wait_asynccnt 0" ::: "memory");
    }
    __syncthreads();

    // ---- build A fragments (16 rows x 256 K) in registers, bf16; exact fp32 xsq ----
    const int rowBase = blockIdx.x * ROWS_PER_WG + wave * 16;
    const int sel = hf * 8;  // lanes 0-15: K {0..7,16..23}; lanes 16-31: K {8..15,24..31}
    const float* xrow = X + (size_t)(rowBase + l16) * D_DIM;

    v16bf A[8];
    float xsq_part = 0.f;
    #pragma unroll
    for (int kk = 0; kk < 8; ++kk) {
        const float4* p0 = (const float4*)(xrow + kk*32 + sel);
        const float4* p1 = (const float4*)(xrow + kk*32 + 16 + sel);
        float4 f0 = p0[0], f1 = p0[1];
        float4 f2 = p1[0], f3 = p1[1];
        xsq_part += f0.x*f0.x + f0.y*f0.y + f0.z*f0.z + f0.w*f0.w;
        xsq_part += f1.x*f1.x + f1.y*f1.y + f1.z*f1.z + f1.w*f1.w;
        xsq_part += f2.x*f2.x + f2.y*f2.y + f2.z*f2.z + f2.w*f2.w;
        xsq_part += f3.x*f3.x + f3.y*f3.y + f3.z*f3.z + f3.w*f3.w;
        v16bf a;
        a[0]=(__bf16)f0.x; a[1]=(__bf16)f0.y; a[2]=(__bf16)f0.z; a[3]=(__bf16)f0.w;
        a[4]=(__bf16)f1.x; a[5]=(__bf16)f1.y; a[6]=(__bf16)f1.z; a[7]=(__bf16)f1.w;
        a[8]=(__bf16)f2.x; a[9]=(__bf16)f2.y; a[10]=(__bf16)f2.z; a[11]=(__bf16)f2.w;
        a[12]=(__bf16)f3.x; a[13]=(__bf16)f3.y; a[14]=(__bf16)f3.z; a[15]=(__bf16)f3.w;
        A[kk] = a;
    }
    // combine halves: every lane gets full ||x||^2 of row rowBase + (lane&15)
    float xsq = xsq_part + __shfl_xor(xsq_part, 16, 32);

    // xsq for the row each acc VGPR j maps to: row = j + hf*8
    // lanes 16-31 read from lane 24+j (upper half holds rows (lane&15)) -> branch-free
    const int xsbase = hf * 24;
    float xs[8];
    #pragma unroll
    for (int j = 0; j < 8; ++j)
        xs[j] = __shfl(xsq, xsbase + j, 32);

    // ---- 16 column tiles x 8 k-steps of v_wmma_f32_16x16x32_bf16 ----
    v8f   qv[16];
    float rowsum[8] = {0.f,0.f,0.f,0.f,0.f,0.f,0.f,0.f};

    #pragma unroll
    for (int t = 0; t < 16; ++t) {
        const __bf16* bbase = Cs + (size_t)(t*16 + l16) * LDS_STRIDE + hf * 8;
        v8f c = {0.f,0.f,0.f,0.f,0.f,0.f,0.f,0.f};
        #pragma unroll
        for (int kk = 0; kk < 8; ++kk) {
            BFrag bf;
            const uint4* bp = (const uint4*)(bbase + kk*32);
            bf.u[0] = bp[0];
            bf.u[1] = bp[2];
            c = __builtin_amdgcn_wmma_f32_16x16x32_bf16(
                    false, A[kk], false, bf.v, (short)0, c, false, false);
        }
        const float cq1 = csq1_s[t*16 + l16];  // ||c||^2 + 1
        v8f q;
        #pragma unroll
        for (int j = 0; j < 8; ++j) {
            // 1 + max(dist2, 0) == max(xs + (csq+1) - 2*cross, 1)
            float d = __builtin_fmaf(-2.0f, c[j], xs[j] + cq1);
            d = d > 1.0f ? d : 1.0f;
            float qq = __builtin_amdgcn_rcpf(d);   // ALPHA = 1 -> exponent is exactly 1
            rowsum[j] += qq;
            q[j] = qq;
        }
        qv[t] = q;
    }

    // ---- row sums: reduce across the 16 lanes of each half (rows j / j+8) ----
    #pragma unroll
    for (int j = 0; j < 8; ++j) {
        float s = rowsum[j];
        s += __shfl_xor(s, 1, 32);
        s += __shfl_xor(s, 2, 32);
        s += __shfl_xor(s, 4, 32);
        s += __shfl_xor(s, 8, 32);
        rowsum[j] = __builtin_amdgcn_rcpf(s);
    }

    // ---- normalize + store (two 64B segments per instruction) ----
    #pragma unroll
    for (int t = 0; t < 16; ++t) {
        const int col = t*16 + l16;
        #pragma unroll
        for (int j = 0; j < 8; ++j) {
            const int row = rowBase + j + hf*8;
            Q[(size_t)row * K_DIM + col] = qv[t][j] * rowsum[j];
        }
    }
}

extern "C" void kernel_launch(void* const* d_in, const int* in_sizes, int n_in,
                              void* d_out, int out_size, void* d_ws, size_t ws_size,
                              hipStream_t stream) {
    const float* X = (const float*)d_in[0];   // (N, 256) fp32
    const float* C = (const float*)d_in[1];   // (256, 256) fp32
    float* Q = (float*)d_out;                 // (N, 256) fp32
    const int N = in_sizes[0] / D_DIM;        // 131072

    // workspace layout: [0,128KB) bf16 C; [128KB, +1KB) csq1
    __bf16* Cbf  = (__bf16*)d_ws;
    float*  csq1 = (float*)((char*)d_ws + (size_t)K_DIM * D_DIM * sizeof(__bf16));

    prep_kernel<<<dim3(1), dim3(256), 0, stream>>>(C, Cbf, csq1);

    const int grid = N / ROWS_PER_WG;         // 1024 workgroups
    cluster_q_kernel<<<dim3(grid), dim3(256), 0, stream>>>(X, Cbf, csq1, Q, N);
}
